// PACA_25125558681911
// MI455X (gfx1250) — matrix operand order repaired and checked by reading, not compiled, not run on hardware
//
#include <hip/hip_runtime.h>
#include <hip/hip_bf16.h>

// ---------------------------------------------------------------------------
// MI455X (gfx1250) implementation. wave32, WMMA 16x16x32 bf16.
// Memory-bound (~17 GFLOP vs ~0.5-1 GB traffic at 23.3 TB/s):
//  - q/k/v bf16, contraction-contiguous per-head layouts (direct fragment loads)
//  - score matrices stored bf16 -> S1+S2 = 134 MB total, resident in 192MB L2
//    across the ori/row-stats/PV passes
//  - rd+polar packed to one u16 table (8.4 MB, L2-resident, gathered ~5x)
//  - probabilities never materialized: recomputed per-lane inside PV kernels
//    directly into the WMMA A-fragment lane/VGPR layout (two bf16x8 runs/lane)
// ---------------------------------------------------------------------------

typedef __bf16 bf16;
typedef __attribute__((ext_vector_type(16))) bf16 bf16x16;
typedef __attribute__((ext_vector_type(8)))  bf16 bf16x8;
typedef __attribute__((ext_vector_type(8)))  float f32x8;
typedef __attribute__((ext_vector_type(8)))  unsigned short u16x8;

#define NB 4
#define NH 8
#define NX 4096
#define NK 256
#define ND 256
#define DH 32
#define SCALE 0.17677669529663687f   // (256/8)^-0.5

__device__ __forceinline__ f32x8 wmma_bf16(bf16x16 a, bf16x16 b, f32x8 c) {
  return __builtin_amdgcn_wmma_f32_16x16x32_bf16(false, a, false, b, (short)0, c,
                                                 false, false);
}

union FragU { bf16x16 w; bf16x8 h[2]; };

// A fragment (16x32): lane L holds row m=L&15; VGPR0-3 = K[8*half .. +7],
// VGPR4-7 = K[16+8*half .. +7] (ISA 7.12.2). src row-major, contraction contig.
__device__ __forceinline__ bf16x16 ldA(const bf16* p, int lda, int lane) {
  int m = lane & 15, half = lane >> 4;
  const bf16* q = p + (size_t)m * lda + 8 * half;
  FragU u;
  u.h[0] = *(const bf16x8*)q;
  u.h[1] = *(const bf16x8*)(q + 16);
  return u.w;
}

// Same but converting from f32 source.
__device__ __forceinline__ bf16x16 ldA_f32(const float* p, int lda, int lane) {
  int m = lane & 15, half = lane >> 4;
  const float* q0 = p + (size_t)m * lda + 8 * half;
  const float* q1 = q0 + 16;
  f32x8 a = *(const f32x8*)q0;
  f32x8 b = *(const f32x8*)q1;
  bf16x16 r;
#pragma unroll
  for (int i = 0; i < 8; ++i) { r[i] = (bf16)a[i]; r[i + 8] = (bf16)b[i]; }
  return r;
}

// B fragment (32x16): lane L holds col n=L&15, K = 8*half .. +7 and 16+8*half .. +7.
// src is "n-major": row = output column, contraction contiguous.
__device__ __forceinline__ bf16x16 ldB(const bf16* p, int ldb, int lane, int kofs) {
  int n = lane & 15, half = lane >> 4;
  const bf16* q = p + (size_t)n * ldb + kofs + 8 * half;
  FragU u;
  u.h[0] = *(const bf16x8*)q;
  u.h[1] = *(const bf16x8*)(q + 16);
  return u.w;
}

// ---------------------------------------------------------------------------
// weight transpose + f32->bf16:  out[c][r] = in[r][c]
__global__ void k_transpose(const float* __restrict__ in, bf16* __restrict__ out,
                            int rows, int cols) {
  int i = blockIdx.x * blockDim.x + threadIdx.x;
  if (i < rows * cols) {
    int r = i / cols, c = i % cols;
    out[(size_t)c * rows + r] = (bf16)in[(size_t)r * cols + c];
  }
}

// pack rd (0..65 -> low byte) + clipped polar (-> high byte) into one u16
__global__ void k_pack(const int* __restrict__ rd, const int* __restrict__ polar,
                       unsigned short* __restrict__ pk, int n) {
  int i = blockIdx.x * blockDim.x + threadIdx.x;
  if (i < n) {
    int p = polar[i]; p = p < 0 ? 0 : (p > 7 ? 7 : p);
    pk[i] = (unsigned short)((rd[i] & 0xff) | (p << 8));
  }
}

// ---------------------------------------------------------------------------
// QKV projection: [NB*rows,256] @ [256,768] -> scatter into per-head layouts
//   q,k : [bh][pos][32] bf16     v : [bh][32][pos] bf16 (transposed)
__global__ void k_qkv(const float* __restrict__ Xin, const bf16* __restrict__ WT,
                      bf16* __restrict__ qout, bf16* __restrict__ kout,
                      bf16* __restrict__ vT, int rows) {
  const int lane = threadIdx.x & 31;
  const int wave = threadIdx.x >> 5;
  const int NGRP = 12;                       // 48 ntiles / 4 per wave
  int wid = blockIdx.x * 8 + wave;
  int mt = wid / NGRP, ng = wid % NGRP;
  int row0 = mt * 16;
  int col0 = ng * 64;

  f32x8 acc[4] = {};
  for (int k0 = 0; k0 < ND; k0 += 32) {
    bf16x16 a = ldA_f32(Xin + (size_t)row0 * ND + k0, ND, lane);
#pragma unroll
    for (int j = 0; j < 4; ++j)
      acc[j] = wmma_bf16(a, ldB(WT + (size_t)(col0 + 16 * j) * ND, ND, lane, k0),
                         acc[j]);
  }

  int bb = row0 / rows;
  int pos_base = row0 - bb * rows;
  int half = lane >> 4;
#pragma unroll
  for (int j = 0; j < 4; ++j) {
    int col = col0 + 16 * j + (lane & 15);
    int which = col >> 8, d0 = col & 255;
    int h = d0 >> 5, dh = d0 & 31;
    size_t bh = (size_t)(bb * NH + h);
#pragma unroll
    for (int r = 0; r < 8; ++r) {
      int pos = pos_base + r + 8 * half;
      bf16 val = (bf16)acc[j][r];
      if (which == 0)      qout[(bh * rows + pos) * DH + dh] = val;
      else if (which == 1) kout[(bh * rows + pos) * DH + dh] = val;
      else                 vT[(bh * DH + dh) * (size_t)rows + pos] = val;
    }
  }
}

// ---------------------------------------------------------------------------
// Scores (bf16 out): S[bh][m][n] = scale * sum_d A[bh][m][d] * Bsrc[bh][n][d]
// contraction = 32 => exactly ONE wmma per 16x16 tile.
__global__ void k_score(const bf16* __restrict__ A, const bf16* __restrict__ Bsrc,
                        bf16* __restrict__ S, int rowsA, int rowsB) {
  int bh = blockIdx.y;
  const bf16* Ab = A + (size_t)bh * rowsA * DH;
  const bf16* Bb = Bsrc + (size_t)bh * rowsB * DH;
  bf16* Sb = S + (size_t)bh * rowsA * rowsB;
  int lane = threadIdx.x & 31, wave = threadIdx.x >> 5;
  int ngrp = rowsB / 64;
  int wid = blockIdx.x * 8 + wave;
  int mt = wid / ngrp, ng = wid % ngrp;
  int row0 = mt * 16, col0 = ng * 64;

  bf16x16 a = ldA(Ab + (size_t)row0 * DH, DH, lane);
  int n = lane & 15, half = lane >> 4;
#pragma unroll
  for (int j = 0; j < 4; ++j) {
    f32x8 c = {};
    c = wmma_bf16(a, ldB(Bb + (size_t)(col0 + 16 * j) * DH, DH, lane, 0), c);
#pragma unroll
    for (int r = 0; r < 8; ++r)
      Sb[(size_t)(row0 + r + 8 * half) * rowsB + col0 + 16 * j + n] =
          (bf16)(c[r] * SCALE);
  }
}

// ---------------------------------------------------------------------------
// main orientation: argmax over 8 polar buckets of sum |S1| along x
__global__ void k_ori(const bf16* __restrict__ S1,
                      const unsigned short* __restrict__ pk,
                      int* __restrict__ ori) {
  int bhk = blockIdx.x;                       // [B*H*K]
  int b = bhk / (NH * NK);
  int k = bhk % NK;
  const bf16* Srow = S1 + (size_t)bhk * NX;
  const unsigned short* pkrow = pk + ((size_t)b * NK + k) * NX;
  float loc[8] = {0, 0, 0, 0, 0, 0, 0, 0};
  for (int x = threadIdx.x; x < NX; x += 256) {
    int p = pkrow[x] >> 8;
    float a = fabsf((float)Srow[x]);
#pragma unroll
    for (int o = 0; o < 8; ++o) loc[o] += (p == o) ? a : 0.0f;
  }
  __shared__ float sh[256 * 8];
#pragma unroll
  for (int o = 0; o < 8; ++o) sh[threadIdx.x * 8 + o] = loc[o];
  __syncthreads();
  for (int s = 128; s > 0; s >>= 1) {
    if (threadIdx.x < s)
#pragma unroll
      for (int o = 0; o < 8; ++o)
        sh[threadIdx.x * 8 + o] += sh[(threadIdx.x + s) * 8 + o];
    __syncthreads();
  }
  if (threadIdx.x == 0) {
    float best = sh[0]; int bi = 0;
    for (int o = 1; o < 8; ++o)
      if (sh[o] > best) { best = sh[o]; bi = o; }   // first-max, like argmax
    ori[bhk] = bi;
  }
}

// ---------------------------------------------------------------------------
// row softmax stats, k-direction (rows over x, len 4096, mask fill -1e6)
__global__ void k_rows1(const bf16* __restrict__ S1,
                        const unsigned char* __restrict__ mask,
                        const unsigned short* __restrict__ pk,
                        const float* __restrict__ dis, const float* __restrict__ pe,
                        const int* __restrict__ ori,
                        float* __restrict__ rmax, float* __restrict__ rsum) {
  int bhk = blockIdx.x;
  int b = bhk / (NH * NK);
  int h = (bhk / NK) % NH;
  int k = bhk % NK;
  const bf16* Srow = S1 + (size_t)bhk * NX;
  const unsigned char* mrow = mask + ((size_t)(b * NH + h) * NX) * NK + k;
  const unsigned short* pkrow = pk + ((size_t)b * NK + k) * NX;
  int o = ori[bhk];

  __shared__ float lbuf[NX];     // 16 KB of the 320 KB WGP LDS
  __shared__ float red[256];
  float mx = -3.4e38f;
  for (int x = threadIdx.x; x < NX; x += 256) {
    float s = (float)Srow[x];
    if (mrow[(size_t)x * NK]) s = -1e6f;
    int pv = pkrow[x];
    int np = (pv >> 8) - o; if (np < 0) np += 8;
    float lg = s + dis[(pv & 0xff) * NH + h] + pe[np];
    lbuf[x] = lg;
    mx = fmaxf(mx, lg);
  }
  red[threadIdx.x] = mx; __syncthreads();
  for (int s = 128; s > 0; s >>= 1) {
    if (threadIdx.x < s) red[threadIdx.x] = fmaxf(red[threadIdx.x], red[threadIdx.x + s]);
    __syncthreads();
  }
  mx = red[0]; __syncthreads();
  float sm = 0.f;
  for (int x = threadIdx.x; x < NX; x += 256) sm += __expf(lbuf[x] - mx);
  red[threadIdx.x] = sm; __syncthreads();
  for (int s = 128; s > 0; s >>= 1) {
    if (threadIdx.x < s) red[threadIdx.x] += red[threadIdx.x + s];
    __syncthreads();
  }
  if (threadIdx.x == 0) { rmax[bhk] = mx; rsum[bhk] = red[0]; }
}

// row softmax stats, x-direction (rows over k, len 256, mask fill -1e9)
__global__ void k_rows2(const bf16* __restrict__ S2,
                        const unsigned char* __restrict__ mask,
                        const unsigned short* __restrict__ pk,
                        const float* __restrict__ dis, const float* __restrict__ pe,
                        const int* __restrict__ ori,
                        float* __restrict__ rmax, float* __restrict__ rsum) {
  int bhx = blockIdx.x;                        // [B*H*X]
  int b = bhx / (NH * NX);
  int h = (bhx / NX) % NH;
  int x = bhx % NX;
  int bh = b * NH + h;
  int kk = threadIdx.x;                        // 0..255
  float s = (float)S2[(size_t)bhx * NK + kk];
  if (mask[((size_t)bh * NX + x) * NK + kk]) s = -1e9f;
  int pv = pk[((size_t)b * NK + kk) * NX + x];
  int o = ori[bh * NK + kk];
  int np = (pv >> 8) - o; if (np < 0) np += 8;
  float lg = s + dis[(pv & 0xff) * NH + h] + pe[np];

  __shared__ float red[256];
  red[kk] = lg; __syncthreads();
  for (int st = 128; st > 0; st >>= 1) {
    if (kk < st) red[kk] = fmaxf(red[kk], red[kk + st]);
    __syncthreads();
  }
  float mx = red[0]; __syncthreads();
  red[kk] = __expf(lg - mx); __syncthreads();
  for (int st = 128; st > 0; st >>= 1) {
    if (kk < st) red[kk] += red[kk + st];
    __syncthreads();
  }
  if (kk == 0) { rmax[bhx] = mx; rsum[bhx] = red[0]; }
}

// ---------------------------------------------------------------------------
// Fused prob + P@V, k-direction: O[k][d] = sum_x softmax(..)[k][x] * v[x][d]
// probs computed per-lane straight into the WMMA A-fragment layout; each
// lane's 16 elements are two contiguous 8-runs -> 16B vector loads.
__global__ void k_pv1(const bf16* __restrict__ S1, const bf16* __restrict__ vT,
                      const unsigned char* __restrict__ mask,
                      const unsigned short* __restrict__ pk,
                      const float* __restrict__ dis, const float* __restrict__ pe,
                      const int* __restrict__ ori, const float* __restrict__ rmax,
                      const float* __restrict__ rsum, float* __restrict__ outf) {
  int lane = threadIdx.x & 31, wave = threadIdx.x >> 5;
  int wid = blockIdx.x * 8 + wave;             // B*H * 16 ktiles = 512 waves
  int bh = wid / 16, kt = wid % 16;
  int b = bh / NH, h = bh % NH;
  int m = lane & 15, half = lane >> 4;
  int krow = kt * 16 + m;
  size_t bhk = (size_t)bh * NK + krow;
  const bf16* Srow = S1 + bhk * NX;
  const unsigned short* pkrow = pk + ((size_t)b * NK + krow) * NX;
  const unsigned char* mcol = mask + ((size_t)bh * NX) * NK + krow;
  int o = ori[bhk];
  float mx = rmax[bhk], inv = 1.0f / rsum[bhk];

  f32x8 acc0 = {}, acc1 = {};
  for (int x0 = 0; x0 < NX; x0 += 32) {
    int xa = x0 + 8 * half;         // lane's run for fragment elements 0..7
    int xb = xa + 16;               // run for elements 8..15
    bf16x8 s0 = *(const bf16x8*)(Srow + xa);
    bf16x8 s1 = *(const bf16x8*)(Srow + xb);
    u16x8 q0 = *(const u16x8*)(pkrow + xa);
    u16x8 q1 = *(const u16x8*)(pkrow + xb);
    bf16x16 afr;
#pragma unroll
    for (int e = 0; e < 8; ++e) {
      float s = (float)s0[e];
      if (mcol[(size_t)(xa + e) * NK]) s = -1e6f;
      int pv = q0[e];
      int np = (pv >> 8) - o; if (np < 0) np += 8;
      float lg = s + dis[(pv & 0xff) * NH + h] + pe[np];
      afr[e] = (bf16)(__expf(lg - mx) * inv);

      float s2 = (float)s1[e];
      if (mcol[(size_t)(xb + e) * NK]) s2 = -1e6f;
      int pv2 = q1[e];
      int np2 = (pv2 >> 8) - o; if (np2 < 0) np2 += 8;
      float lg2 = s2 + dis[(pv2 & 0xff) * NH + h] + pe[np2];
      afr[e + 8] = (bf16)(__expf(lg2 - mx) * inv);
    }
    acc0 = wmma_bf16(afr, ldB(vT + (size_t)bh * DH * NX, NX, lane, x0), acc0);
    acc1 = wmma_bf16(afr, ldB(vT + ((size_t)bh * DH + 16) * NX, NX, lane, x0), acc1);
  }
  int n = lane & 15;
#pragma unroll
  for (int r = 0; r < 8; ++r) {
    int kr = kt * 16 + r + 8 * half;
    outf[((size_t)(b * NK + kr)) * ND + h * DH + n]      = acc0[r];
    outf[((size_t)(b * NK + kr)) * ND + h * DH + 16 + n] = acc1[r];
  }
}

// Fused prob + P@V, x-direction: O[x][d] = sum_k softmax(..)[x][k] * kv[k][d]
__global__ void k_pv2(const bf16* __restrict__ S2, const bf16* __restrict__ kvT,
                      const unsigned char* __restrict__ mask,
                      const unsigned short* __restrict__ pk,
                      const float* __restrict__ dis, const float* __restrict__ pe,
                      const int* __restrict__ ori, const float* __restrict__ rmax,
                      const float* __restrict__ rsum, float* __restrict__ outf) {
  int lane = threadIdx.x & 31, wave = threadIdx.x >> 5;
  int wid = blockIdx.x * 8 + wave;             // B*H * 256 xtiles = 8192 waves
  int bh = wid / 256, xt = wid % 256;
  int b = bh / NH, h = bh % NH;
  int m = lane & 15, half = lane >> 4;
  int xrow = xt * 16 + m;
  size_t bhx = (size_t)bh * NX + xrow;
  const bf16* Srow = S2 + bhx * NK;
  const unsigned char* mrow = mask + bhx * NK;
  float mx = rmax[bhx], inv = 1.0f / rsum[bhx];

  f32x8 acc0 = {}, acc1 = {};
  for (int k0 = 0; k0 < NK; k0 += 32) {
    int ka = k0 + 8 * half;
    int kb = ka + 16;
    bf16x8 s0 = *(const bf16x8*)(Srow + ka);
    bf16x8 s1 = *(const bf16x8*)(Srow + kb);
    bf16x16 afr;
#pragma unroll
    for (int e = 0; e < 8; ++e) {
      {
        int kk = ka + e;
        float s = (float)s0[e];
        if (mrow[kk]) s = -1e9f;
        int pv = pk[((size_t)b * NK + kk) * NX + xrow];
        int o = ori[bh * NK + kk];
        int np = (pv >> 8) - o; if (np < 0) np += 8;
        float lg = s + dis[(pv & 0xff) * NH + h] + pe[np];
        afr[e] = (bf16)(__expf(lg - mx) * inv);
      }
      {
        int kk = kb + e;
        float s = (float)s1[e];
        if (mrow[kk]) s = -1e9f;
        int pv = pk[((size_t)b * NK + kk) * NX + xrow];
        int o = ori[bh * NK + kk];
        int np = (pv >> 8) - o; if (np < 0) np += 8;
        float lg = s + dis[(pv & 0xff) * NH + h] + pe[np];
        afr[e + 8] = (bf16)(__expf(lg - mx) * inv);
      }
    }
    acc0 = wmma_bf16(afr, ldB(kvT + (size_t)bh * DH * NK, NK, lane, k0), acc0);
    acc1 = wmma_bf16(afr, ldB(kvT + ((size_t)bh * DH + 16) * NK, NK, lane, k0), acc1);
  }
  int n = lane & 15;
#pragma unroll
  for (int r = 0; r < 8; ++r) {
    int xr = xt * 16 + r + 8 * half;
    outf[((size_t)(b * NX + xr)) * ND + h * DH + n]      = acc0[r];
    outf[((size_t)(b * NX + xr)) * ND + h * DH + 16 + n] = acc1[r];
  }
}

// ---------------------------------------------------------------------------
// final projection: out[m][n] = A[m][:]@w_proj[:,n] + b[n]   (WT is n-major bf16)
__global__ void k_proj(const float* __restrict__ Ain, const bf16* __restrict__ WT,
                       const float* __restrict__ bias, float* __restrict__ out) {
  int lane = threadIdx.x & 31, wave = threadIdx.x >> 5;
  int wid = blockIdx.x * 8 + wave;
  int mt = wid / 4, ng = wid % 4;              // 16 ntiles / 4 per wave
  int row0 = mt * 16, col0 = ng * 64;
  f32x8 acc[4] = {};
  for (int k0 = 0; k0 < ND; k0 += 32) {
    bf16x16 a = ldA_f32(Ain + (size_t)row0 * ND + k0, ND, lane);
#pragma unroll
    for (int j = 0; j < 4; ++j)
      acc[j] = wmma_bf16(a, ldB(WT + (size_t)(col0 + 16 * j) * ND, ND, lane, k0),
                         acc[j]);
  }
  int n = lane & 15, half = lane >> 4;
#pragma unroll
  for (int j = 0; j < 4; ++j) {
    float bi = bias[col0 + 16 * j + n];
#pragma unroll
    for (int r = 0; r < 8; ++r)
      out[(size_t)(row0 + r + 8 * half) * ND + col0 + 16 * j + n] = acc[j][r] + bi;
  }
}

// ---------------------------------------------------------------------------
extern "C" void kernel_launch(void* const* d_in, const int* in_sizes, int n_in,
                              void* d_out, int out_size, void* d_ws, size_t ws_size,
                              hipStream_t stream) {
  const float* x         = (const float*)d_in[0];   // [4,4096,256]
  const float* kern      = (const float*)d_in[1];   // [4,256,256]
  const int*   rd        = (const int*)d_in[2];     // [4,256,4096]
  const int*   polar     = (const int*)d_in[3];     // [4,256,4096]
  const unsigned char* mask = (const unsigned char*)d_in[4]; // [4,8,4096,256] bool
  const float* w_qkv     = (const float*)d_in[5];   // [256,768]
  const float* w_proj    = (const float*)d_in[6];   // [256,256]
  const float* b_proj    = (const float*)d_in[7];   // [256]
  const float* polar_emb = (const float*)d_in[8];   // [8,1]
  const float* dis_emb   = (const float*)d_in[9];   // [66,8]
  float* out = (float*)d_out;                       // x_out (4194304) ++ k_out (262144)

  char* ws = (char*)d_ws;
  size_t off = 0;
  auto alloc = [&](size_t bytes) -> void* {
    void* p = ws + off;
    off = (off + bytes + 255) & ~(size_t)255;
    return p;
  };
  bf16* q_x   = (bf16*)alloc((size_t)NB * NH * NX * DH * 2);
  bf16* k_x   = (bf16*)alloc((size_t)NB * NH * NX * DH * 2);
  bf16* vT_x  = (bf16*)alloc((size_t)NB * NH * NX * DH * 2);
  bf16* q_k   = (bf16*)alloc((size_t)NB * NH * NK * DH * 2);
  bf16* k_k   = (bf16*)alloc((size_t)NB * NH * NK * DH * 2);
  bf16* vT_k  = (bf16*)alloc((size_t)NB * NH * NK * DH * 2);
  bf16* wqkvT = (bf16*)alloc((size_t)768 * 256 * 2);
  bf16* wprT  = (bf16*)alloc((size_t)256 * 256 * 2);
  bf16* S1    = (bf16*)alloc((size_t)NB * NH * NK * NX * 2);  // 67 MB (L2-resident)
  bf16* S2    = (bf16*)alloc((size_t)NB * NH * NX * NK * 2);  // 67 MB
  unsigned short* pk = (unsigned short*)alloc((size_t)NB * NK * NX * 2); // 8.4 MB
  int*   ori  = (int*)alloc((size_t)NB * NH * NK * 4);
  float* rmax1 = (float*)alloc((size_t)NB * NH * NK * 4);
  float* rsum1 = (float*)alloc((size_t)NB * NH * NK * 4);
  float* rmax2 = (float*)alloc((size_t)NB * NH * NX * 4);
  float* rsum2 = (float*)alloc((size_t)NB * NH * NX * 4);
  float* kout_f = (float*)alloc((size_t)NB * NK * ND * 4);
  float* xout_f = (float*)alloc((size_t)NB * NX * ND * 4);

  // 1) weights -> bf16 n-major; rd/polar -> packed u16
  k_transpose<<<(256 * 768 + 255) / 256, 256, 0, stream>>>(w_qkv, wqkvT, 256, 768);
  k_transpose<<<(256 * 256 + 255) / 256, 256, 0, stream>>>(w_proj, wprT, 256, 256);
  k_pack<<<(NB * NK * NX + 255) / 256, 256, 0, stream>>>(rd, polar, pk,
                                                         NB * NK * NX);

  // 2) QKV projections (WMMA). M=16384 -> 1536 blocks; M=1024 -> 96 blocks.
  k_qkv<<<1536, 256, 0, stream>>>(x, wqkvT, q_x, k_x, vT_x, NX);
  k_qkv<<<96, 256, 0, stream>>>(kern, wqkvT, q_k, k_k, vT_k, NK);

  // 3) score matrices (WMMA, contraction = 32), bf16 output
  k_score<<<dim3(128, NB * NH), 256, 0, stream>>>(q_k, k_x, S1, NK, NX);
  k_score<<<dim3(128, NB * NH), 256, 0, stream>>>(q_x, k_k, S2, NX, NK);

  // 4) polar main-orientation argmax (per b,h,k)
  k_ori<<<NB * NH * NK, 256, 0, stream>>>(S1, pk, ori);

  // 5) softmax row stats
  k_rows1<<<NB * NH * NK, 256, 0, stream>>>(S1, mask, pk, dis_emb, polar_emb,
                                            ori, rmax1, rsum1);
  k_rows2<<<NB * NH * NX, 256, 0, stream>>>(S2, mask, pk, dis_emb, polar_emb,
                                            ori, rmax2, rsum2);

  // 6) fused probability + P@V (WMMA)
  k_pv1<<<64, 256, 0, stream>>>(S1, vT_x, mask, pk, dis_emb, polar_emb,
                                ori, rmax1, rsum1, kout_f);
  k_pv2<<<1024, 256, 0, stream>>>(S2, vT_k, mask, pk, dis_emb, polar_emb,
                                  ori, rmax2, rsum2, xout_f);

  // 7) output projection + bias (WMMA). x_out first, then k_out.
  k_proj<<<512, 256, 0, stream>>>(xout_f, wprT, b_proj, out);
  k_proj<<<32, 256, 0, stream>>>(kout_f, wprT, b_proj, out + (size_t)NB * NX * ND);
}
